// SelfAttention_33200097198640
// MI455X (gfx1250) — compile-verified, matrix operand order checked by reading, not run on hardware
//
#include <hip/hip_runtime.h>
#include <hip/hip_bf16.h>

typedef __bf16 bf16_t;
typedef __bf16 v16bf __attribute__((ext_vector_type(16)));
typedef __bf16 v8bf  __attribute__((ext_vector_type(8)));
typedef float  v8f   __attribute__((ext_vector_type(8)));

union BF16Frag { v16bf v; v8bf h[2]; };

#define EMBED 1024
#define SEQ   2048
#define BATCH 2
#define NHEAD 16
#define DHEAD 64
#define MTOT  (BATCH*SEQ)   // 4096 rows

// ---------------------------------------------------------------------------
// 16-lane butterfly reductions in pure VALU via DPP ROW_XMASK (no LDS traffic;
// co-executes with XDL WMMA). xor masks 1/2/4/8 stay within a 16-lane row.
// ---------------------------------------------------------------------------
template<int MASK>
__device__ __forceinline__ float dpp_xor16(float v) {
    int s = __builtin_bit_cast(int, v);
    int r = __builtin_amdgcn_update_dpp(s, s, 0x160 | MASK, 0xf, 0xf, true);
    return __builtin_bit_cast(float, r);
}
__device__ __forceinline__ float rowmax16(float v) {
    v = fmaxf(v, dpp_xor16<1>(v));
    v = fmaxf(v, dpp_xor16<2>(v));
    v = fmaxf(v, dpp_xor16<4>(v));
    v = fmaxf(v, dpp_xor16<8>(v));
    return v;
}
__device__ __forceinline__ float rowsum16(float v) {
    v += dpp_xor16<1>(v);
    v += dpp_xor16<2>(v);
    v += dpp_xor16<4>(v);
    v += dpp_xor16<8>(v);
    return v;
}

// ---------------------------------------------------------------------------
// Stage 0: f32 -> bf16 conversion (8 elems / thread, vector load+store)
// ---------------------------------------------------------------------------
__global__ __launch_bounds__(256) void cvt_f32_bf16(const float* __restrict__ in,
                                                    bf16_t* __restrict__ out, int n8) {
    int i = blockIdx.x * 256 + threadIdx.x;
    if (i < n8) {
        const float4* p = (const float4*)in + (size_t)i * 2;
        float4 a = p[0], b = p[1];
        v8bf r;
        r[0] = (bf16_t)a.x; r[1] = (bf16_t)a.y; r[2] = (bf16_t)a.z; r[3] = (bf16_t)a.w;
        r[4] = (bf16_t)b.x; r[5] = (bf16_t)b.y; r[6] = (bf16_t)b.z; r[7] = (bf16_t)b.w;
        *(v8bf*)(out + (size_t)i * 8) = r;
    }
}

// ---------------------------------------------------------------------------
// GEMM: Y[m,f] = sum_e A[m,e] * W[f,e] + bias[f]
// Block tile 128(M) x 64(N), 256 threads = 8 waves, each wave 32x32 (2x2 wmma).
// Double-buffered LDS with register staging (one barrier per K-step).
// MODE 0: bf16 out in [B*H, S, D]     (Q, K for attention)
// MODE 1: bf16 out in [B*H, D, S]     (V transposed -> packed b128 stores)
// MODE 2: f32  out row-major [M, E]   (final output)
// ---------------------------------------------------------------------------
template<int MODE>
__global__ __launch_bounds__(256) void gemm_bias(const bf16_t* __restrict__ A,
                                                 const bf16_t* __restrict__ W,
                                                 const float*  __restrict__ bias,
                                                 void* __restrict__ outp) {
    constexpr int AST = 40;           // padded LDS row strides (bank stagger)
    constexpr int BST = 40;
    __shared__ bf16_t as[2][128 * AST];
    __shared__ bf16_t bs[2][64 * BST];

    const int tid  = threadIdx.x;
    const int lane = tid & 31;
    const int wave = tid >> 5;
    const int wm   = wave >> 1;       // 0..3
    const int wn   = wave & 1;        // 0..1
    const int mloc = lane & 15;
    const int half = lane >> 4;
    const int m0   = blockIdx.x * 128;
    const int n0   = blockIdx.y * 64;

    // chunk assignment: A = 512 chunks of 8 bf16 (2/thread), B = 256 (1/thread)
    int aRow[2], aKof[2];
#pragma unroll
    for (int cc = 0; cc < 2; ++cc) {
        int c = tid + cc * 256;
        aRow[cc] = c >> 2;
        aKof[cc] = (c & 3) * 8;
    }
    const int bRow = tid >> 2, bKof = (tid & 3) * 8;

    v8bf ra[2], rb;
    auto gload = [&](int k0) {
#pragma unroll
        for (int cc = 0; cc < 2; ++cc)
            ra[cc] = *(const v8bf*)&A[(size_t)(m0 + aRow[cc]) * EMBED + k0 + aKof[cc]];
        rb = *(const v8bf*)&W[(size_t)(n0 + bRow) * EMBED + k0 + bKof];
    };
    auto sstore = [&](int b) {
#pragma unroll
        for (int cc = 0; cc < 2; ++cc)
            *(v8bf*)&as[b][aRow[cc] * AST + aKof[cc]] = ra[cc];
        *(v8bf*)&bs[b][bRow * BST + bKof] = rb;
    };

    v8f acc[2][2] = {};

    gload(0);
    sstore(0);
    __syncthreads();

    constexpr int NK = EMBED / 32;
    for (int kt = 0; kt < NK; ++kt) {
        const int  cur  = kt & 1;
        const bool more = (kt + 1) < NK;
        if (more) gload((kt + 1) * 32);     // global loads overlap the wmma below

        BF16Frag af[2], bfm[2];
#pragma unroll
        for (int mi = 0; mi < 2; ++mi) {
            int r = wm * 32 + mi * 16 + mloc;
            af[mi].h[0] = *(const v8bf*)&as[cur][r * AST + half * 8];
            af[mi].h[1] = *(const v8bf*)&as[cur][r * AST + 16 + half * 8];
        }
#pragma unroll
        for (int ni = 0; ni < 2; ++ni) {
            int r = wn * 32 + ni * 16 + mloc;
            bfm[ni].h[0] = *(const v8bf*)&bs[cur][r * BST + half * 8];
            bfm[ni].h[1] = *(const v8bf*)&bs[cur][r * BST + 16 + half * 8];
        }
#pragma unroll
        for (int mi = 0; mi < 2; ++mi)
#pragma unroll
            for (int ni = 0; ni < 2; ++ni)
                acc[mi][ni] = __builtin_amdgcn_wmma_f32_16x16x32_bf16(
                    false, af[mi].v, false, bfm[ni].v,
                    (short)0, acc[mi][ni], false, false);

        if (more) sstore(cur ^ 1);
        __syncthreads();
    }

    // Epilogue. D layout: lane holds column n=mloc; vgpr j holds row j+8*half.
#pragma unroll
    for (int mi = 0; mi < 2; ++mi) {
#pragma unroll
        for (int ni = 0; ni < 2; ++ni) {
            const int   col = n0 + wn * 32 + ni * 16 + mloc;
            const float bv  = bias[col];
            if (MODE == 1) {
                // rows j are consecutive s -> pack 8 bf16 into one b128 store
                const int    h  = col >> 6, d = col & (DHEAD - 1);
                const int    r0 = m0 + wm * 32 + mi * 16 + 8 * half;  // s of j=0
                const int    b  = r0 >> 11, s0 = r0 & (SEQ - 1);
                v8bf r;
#pragma unroll
                for (int j = 0; j < 8; ++j) r[j] = (bf16_t)(acc[mi][ni][j] + bv);
                *(v8bf*)&((bf16_t*)outp)[(((size_t)(b * NHEAD + h)) * DHEAD + d) * SEQ + s0] = r;
            } else {
#pragma unroll
                for (int j = 0; j < 8; ++j) {
                    const int   row = m0 + wm * 32 + mi * 16 + j + 8 * half;
                    const float v   = acc[mi][ni][j] + bv;
                    if (MODE == 0) {
                        int b = row >> 11, s = row & (SEQ - 1);
                        int h = col >> 6,  d = col & (DHEAD - 1);
                        ((bf16_t*)outp)[(((size_t)(b * NHEAD + h)) * SEQ + s) * DHEAD + d] =
                            (bf16_t)v;
                    } else {
                        ((float*)outp)[(size_t)row * EMBED + col] = v;
                    }
                }
            }
        }
    }
}

// ---------------------------------------------------------------------------
// Flash-style attention. Q,K in [B*H, S, D]; V pre-transposed [B*H, D, S].
// 128 threads (4 waves), 64 query rows per block, key blocks of 64.
// Double-buffered K/V^T staging; softmax reductions via DPP; P transposed
// through a per-wave LDS slab. Output ctx in [B, S, E] bf16.
// ---------------------------------------------------------------------------
__global__ __launch_bounds__(128) void attn(const bf16_t* __restrict__ Q,
                                            const bf16_t* __restrict__ K,
                                            const bf16_t* __restrict__ Vt,
                                            bf16_t* __restrict__ ctx) {
    constexpr int KB  = 64;   // keys per block iteration
    constexpr int KST = 72;   // padded LDS strides
    constexpr int VST = 72;
    constexpr int PST = 72;
    __shared__ bf16_t ks[2][KB * KST];     // [key][e]
    __shared__ bf16_t vs[2][DHEAD * VST];  // [d][key]
    __shared__ bf16_t plds[4][16 * PST];   // per-wave P transpose slab

    const int tid  = threadIdx.x;
    const int lane = tid & 31;
    const int wave = tid >> 5;
    const int mloc = lane & 15;
    const int half = lane >> 4;
    const int bh   = blockIdx.y;
    const int q0   = blockIdx.x * 64 + wave * 16;

    // Q fragments for this wave's 16 rows; fold in 1/sqrt(D) once.
    const bf16_t* Qb = Q + ((size_t)bh * SEQ + q0 + mloc) * DHEAD;
    BF16Frag aq[2];
#pragma unroll
    for (int kc = 0; kc < 2; ++kc) {
        aq[kc].h[0] = *(const v8bf*)&Qb[kc * 32 + half * 8];
        aq[kc].h[1] = *(const v8bf*)&Qb[kc * 32 + 16 + half * 8];
#pragma unroll
        for (int e = 0; e < 16; ++e)
            aq[kc].v[e] = (bf16_t)((float)aq[kc].v[e] * 0.125f);
    }

    // staging chunk assignment: 512 chunks of 8 bf16, 4 per thread
    int cRow[4], cKof[4];
#pragma unroll
    for (int cc = 0; cc < 4; ++cc) {
        int c = tid + cc * 128;
        cRow[cc] = c >> 3;
        cKof[cc] = (c & 7) * 8;
    }
    const bf16_t* Kbase = K  + (size_t)bh * SEQ * DHEAD;
    const bf16_t* Vbase = Vt + (size_t)bh * DHEAD * SEQ;   // [d][s]

    v8bf rk[4], rv[4];
    auto gload = [&](int kb) {
        const bf16_t* Kg = Kbase + (size_t)kb * KB * DHEAD;
#pragma unroll
        for (int cc = 0; cc < 4; ++cc)
            rk[cc] = *(const v8bf*)&Kg[cRow[cc] * DHEAD + cKof[cc]];
        const bf16_t* Vg = Vbase + kb * KB;                // row d, keys kb*KB..
#pragma unroll
        for (int cc = 0; cc < 4; ++cc)
            rv[cc] = *(const v8bf*)&Vg[(size_t)cRow[cc] * SEQ + cKof[cc]];
        if (kb + 1 < SEQ / KB) {   // prefetch one more block ahead, near caches
            __builtin_prefetch((const char*)(Kg + KB * DHEAD) + tid * 64, 0, 3);
            __builtin_prefetch((const char*)(Vg + KB) + (tid >> 1) * ((size_t)SEQ * 2), 0, 3);
        }
    };
    auto sstore = [&](int b) {
#pragma unroll
        for (int cc = 0; cc < 4; ++cc)
            *(v8bf*)&ks[b][cRow[cc] * KST + cKof[cc]] = rk[cc];
#pragma unroll
        for (int cc = 0; cc < 4; ++cc)
            *(v8bf*)&vs[b][cRow[cc] * VST + cKof[cc]] = rv[cc];  // already [d][key]
    };

    v8f   o[4] = {};
    float mrow[8], lrow[8];
#pragma unroll
    for (int j = 0; j < 8; ++j) { mrow[j] = -1e30f; lrow[j] = 0.0f; }

    bf16_t* pb = &plds[wave][0];

    gload(0);
    sstore(0);
    __syncthreads();

    constexpr int NKB = SEQ / KB;
    for (int kb = 0; kb < NKB; ++kb) {
        const int  cur  = kb & 1;
        const bool more = (kb + 1) < NKB;
        if (more) gload(kb + 1);           // overlap global latency with wmma

        // scores: 16x64 tile = 4 n-tiles, each over K=64 (2 wmma)
        v8f sacc[4] = {};
#pragma unroll
        for (int nt = 0; nt < 4; ++nt) {
#pragma unroll
            for (int kc = 0; kc < 2; ++kc) {
                BF16Frag bk;
                const int r = nt * 16 + mloc;  // key row
                bk.h[0] = *(const v8bf*)&ks[cur][r * KST + kc * 32 + half * 8];
                bk.h[1] = *(const v8bf*)&ks[cur][r * KST + kc * 32 + 16 + half * 8];
                sacc[nt] = __builtin_amdgcn_wmma_f32_16x16x32_bf16(
                    false, aq[kc].v, false, bk.v, (short)0, sacc[nt], false, false);
            }
        }

        // online softmax per query row j+8*half (row spans 16 lanes of one half)
        float p[4][8];
#pragma unroll
        for (int j = 0; j < 8; ++j) {
            float rm = fmaxf(fmaxf(sacc[0][j], sacc[1][j]),
                             fmaxf(sacc[2][j], sacc[3][j]));
            rm = rowmax16(rm);
            const float mn = fmaxf(mrow[j], rm);
            const float sf = __expf(mrow[j] - mn);
            float sum = 0.0f;
#pragma unroll
            for (int nt = 0; nt < 4; ++nt) {
                p[nt][j] = __expf(sacc[nt][j] - mn);
                sum += p[nt][j];
            }
            sum     = rowsum16(sum);
            lrow[j] = lrow[j] * sf + sum;
            mrow[j] = mn;
#pragma unroll
            for (int dt = 0; dt < 4; ++dt) o[dt][j] = o[dt][j] * sf;
        }

        // transpose P (D-layout -> A-layout) via this wave's private LDS slab
#pragma unroll
        for (int nt = 0; nt < 4; ++nt)
#pragma unroll
            for (int j = 0; j < 8; ++j)
                pb[(j + 8 * half) * PST + nt * 16 + mloc] = (bf16_t)p[nt][j];
        asm volatile("s_wait_dscnt 0" ::: "memory");   // LDS in-order per wave

        BF16Frag pf[2];
#pragma unroll
        for (int kc = 0; kc < 2; ++kc) {
            pf[kc].h[0] = *(const v8bf*)&pb[mloc * PST + kc * 32 + half * 8];
            pf[kc].h[1] = *(const v8bf*)&pb[mloc * PST + kc * 32 + 16 + half * 8];
        }

        // O += P(16x64) * V(64x64)
#pragma unroll
        for (int dt = 0; dt < 4; ++dt) {
#pragma unroll
            for (int kc = 0; kc < 2; ++kc) {
                BF16Frag bv;
                const int r = dt * 16 + mloc;  // d column
                bv.h[0] = *(const v8bf*)&vs[cur][r * VST + kc * 32 + half * 8];
                bv.h[1] = *(const v8bf*)&vs[cur][r * VST + kc * 32 + 16 + half * 8];
                o[dt] = __builtin_amdgcn_wmma_f32_16x16x32_bf16(
                    false, pf[kc].v, false, bv.v, (short)0, o[dt], false, false);
            }
        }

        if (more) sstore(cur ^ 1);
        __syncthreads();
    }

    // normalize and write ctx as [B, S, E] (E = h*64 + d), bf16
    const int b = bh >> 4, h = bh & 15;
#pragma unroll
    for (int j = 0; j < 8; ++j) {
        const float  inv  = 1.0f / lrow[j];
        const int    row  = q0 + j + 8 * half;
        const size_t base = ((size_t)b * SEQ + row) * EMBED + h * DHEAD;
#pragma unroll
        for (int dt = 0; dt < 4; ++dt)
            ctx[base + dt * 16 + mloc] = (bf16_t)(o[dt][j] * inv);
    }
}

// ---------------------------------------------------------------------------
// Host-side orchestration
// ---------------------------------------------------------------------------
extern "C" void kernel_launch(void* const* d_in, const int* in_sizes, int n_in,
                              void* d_out, int out_size, void* d_ws, size_t ws_size,
                              hipStream_t stream) {
    (void)in_sizes; (void)n_in; (void)out_size; (void)ws_size;

    const float* x  = (const float*)d_in[0];
    const float* Wq = (const float*)d_in[1];
    const float* bq = (const float*)d_in[2];
    const float* Wk = (const float*)d_in[3];
    const float* bk = (const float*)d_in[4];
    const float* Wv = (const float*)d_in[5];
    const float* bv = (const float*)d_in[6];
    const float* Wo = (const float*)d_in[7];
    const float* bo = (const float*)d_in[8];
    float* out = (float*)d_out;

    char* ws = (char*)d_ws;
    bf16_t* xb  = (bf16_t*)(ws);                          // 4M elems (8 MB)
    bf16_t* wqb = (bf16_t*)(ws + (size_t)8 * 1024 * 1024);
    bf16_t* wkb = wqb + 1024 * 1024;
    bf16_t* wvb = wkb + 1024 * 1024;
    bf16_t* wob = wvb + 1024 * 1024;
    bf16_t* Qb  = wob + 1024 * 1024;                      // [B*H, S, D]
    bf16_t* Kb  = Qb + (size_t)4 * 1024 * 1024;           // [B*H, S, D]
    bf16_t* Vb  = Kb + (size_t)4 * 1024 * 1024;           // [B*H, D, S] (transposed)
    bf16_t* Cb  = Vb + (size_t)4 * 1024 * 1024;           // ctx [B, S, E]

    const int NX = MTOT * EMBED;      // 4194304
    const int NW = EMBED * EMBED;     // 1048576
    cvt_f32_bf16<<<(NX / 8 + 255) / 256, 256, 0, stream>>>(x, xb, NX / 8);
    cvt_f32_bf16<<<(NW / 8 + 255) / 256, 256, 0, stream>>>(Wq, wqb, NW / 8);
    cvt_f32_bf16<<<(NW / 8 + 255) / 256, 256, 0, stream>>>(Wk, wkb, NW / 8);
    cvt_f32_bf16<<<(NW / 8 + 255) / 256, 256, 0, stream>>>(Wv, wvb, NW / 8);
    cvt_f32_bf16<<<(NW / 8 + 255) / 256, 256, 0, stream>>>(Wo, wob, NW / 8);

    dim3 gg(MTOT / 128, EMBED / 64);
    gemm_bias<0><<<gg, 256, 0, stream>>>(xb, wqb, bq, Qb);
    gemm_bias<0><<<gg, 256, 0, stream>>>(xb, wkb, bk, Kb);
    gemm_bias<1><<<gg, 256, 0, stream>>>(xb, wvb, bv, Vb);

    attn<<<dim3(SEQ / 64, BATCH * NHEAD), 128, 0, stream>>>(Qb, Kb, Vb, Cb);

    gemm_bias<2><<<gg, 256, 0, stream>>>(Cb, wob, bo, out);
}
